// OrdiPoseNet_50440095924404
// MI455X (gfx1250) — matrix operand order repaired and checked by reading, not compiled, not run on hardware
//
#include <hip/hip_runtime.h>

typedef __attribute__((ext_vector_type(16))) __bf16 v16bf;
typedef __attribute__((ext_vector_type(8)))  __bf16 v8bf;
typedef __attribute__((ext_vector_type(8)))  float  v8f;

#define N_BATCH 1024
#define BDIM    2048
#define LATENT  1024
#define NCLS    64

// ---------------- f32 -> bf16 elementwise convert ----------------
__global__ void cvt_f32_bf16(const float* __restrict__ in, __bf16* __restrict__ out, int n) {
  int i = blockIdx.x * blockDim.x + threadIdx.x;
  if (i < n) out[i] = (__bf16)in[i];
}

// ------------- pack [cls_pos_w; cls_orient_w] -> bf16 [128,1024] -------------
__global__ void pack_clsw(const float* __restrict__ pos, const float* __restrict__ ori,
                          __bf16* __restrict__ out) {
  int i = blockIdx.x * blockDim.x + threadIdx.x;   // 0..131071
  if (i >= 128 * 1024) return;
  int j = i >> 10, k = i & 1023;
  float v = (j < 64) ? pos[j * 1024 + k] : ori[(j - 64) * 1024 + k];
  out[i] = (__bf16)v;
}

// ---------------- 7x7 avg-pool: feat[N,BD,7,7] -> pooled bf16 [N,BD] ----------------
// This is the only HBM-bound stage (~411 MB @ 23.3 TB/s ~= 17.6 us). Use the CDNA5
// async global->LDS path (GLOBAL_LOAD_ASYNC_TO_LDS_B128, tracked by ASYNCcnt):
// data lands directly in LDS with no VGPR round trip; one s_wait_asynccnt 0 +
// workgroup barrier before the 49-wide reductions (stride 49 is odd -> conflict-free
// across the 64 LDS banks).
__global__ void pool_kernel(const float* __restrict__ feat, __bf16* __restrict__ pooled) {
  __shared__ float s[128 * 49];                              // 25 KB of 320 KB/WGP
  const int tid = threadIdx.x;
  // Low 32 bits of a generic LDS pointer == workgroup-relative LDS byte offset.
  const unsigned int lds_base = (unsigned int)(unsigned long long)(&s[0]);
  const unsigned long long gbase =
      (unsigned long long)(const void*)feat +
      (unsigned long long)blockIdx.x * (128ull * 49ull * 4ull);
  for (int i = tid; i < 128 * 49 / 4; i += 256) {            // 1568 x 16B per block
    unsigned int       lds_off = lds_base + (unsigned int)i * 16u;
    unsigned long long gaddr   = gbase + (unsigned long long)i * 16ull;
    asm volatile("global_load_async_to_lds_b128 %0, %1, off"
                 :: "v"(lds_off), "v"(gaddr) : "memory");
  }
  asm volatile("s_wait_asynccnt 0x0" ::: "memory");
  __syncthreads();
  if (tid < 128) {
    float sum = 0.f;
    #pragma unroll
    for (int j = 0; j < 49; ++j) sum += s[tid * 49 + j];
    pooled[(long)blockIdx.x * 128 + tid] = (__bf16)(sum * (1.0f / 49.0f));
  }
}

// ---------------- GEMM1: X = relu(pooled[1024,2048] @ fc1_w^T + b) ----------------
// One 16x16 output tile per wave; K=2048 in 64 steps of v_wmma_f32_16x16x32_bf16.
// A lane layout chosen here (16-bit A 16x32): lane holds row=lane%16, 16 contiguous K starting
// at kb + 16*(lane/16). B lane layout chosen here, same k order as A (32x16): lane holds column
// n=lane%16, 16 contiguous K starting at kb + 16*(lane/16), pairs packed lo/hi.
__global__ void gemm1_wmma(const __bf16* __restrict__ A,   // pooled [1024,2048]
                           const __bf16* __restrict__ W,   // fc1_w bf16 [1024,2048]
                           const float*  __restrict__ bias,
                           float*  __restrict__ Xf,        // [1024,1024] fp32
                           __bf16* __restrict__ Xb) {      // [1024,1024] bf16
  const int lane = threadIdx.x & 31;
  const int wave = threadIdx.x >> 5;
  const int half = lane >> 4;
  const int lr   = lane & 15;
  const int row_base = blockIdx.y * 16;
  const int col_base = (blockIdx.x * 8 + wave) * 16;

  const __bf16* aRow = A + (long)(row_base + lr) * BDIM;
  const __bf16* bRow = W + (long)(col_base + lr) * BDIM;   // B col n = fc1_w row n

  v8f c = {};
  #pragma unroll 4
  for (int kb = 0; kb < BDIM; kb += 32) {
    v8bf a0 = *reinterpret_cast<const v8bf*>(aRow + kb + half * 16);
    v8bf a1 = *reinterpret_cast<const v8bf*>(aRow + kb + half * 16 + 8);
    v8bf b0 = *reinterpret_cast<const v8bf*>(bRow + kb + half * 16);
    v8bf b1 = *reinterpret_cast<const v8bf*>(bRow + kb + half * 16 + 8);
    v16bf av = __builtin_shufflevector(a0, a1, 0,1,2,3,4,5,6,7,8,9,10,11,12,13,14,15);
    v16bf bv = __builtin_shufflevector(b0, b1, 0,1,2,3,4,5,6,7,8,9,10,11,12,13,14,15);
    c = __builtin_amdgcn_wmma_f32_16x16x32_bf16(false, av, false, bv, (short)0, c, false, false);
  }
  const int col = col_base + lr;
  const float bcol = bias[col];
  #pragma unroll
  for (int i = 0; i < 8; ++i) {
    int row = row_base + i + half * 8;
    float v = c[i] + bcol;
    v = v > 0.f ? v : 0.f;
    Xf[(long)row * LATENT + col] = v;
    Xb[(long)row * LATENT + col] = (__bf16)v;
  }
}

// ---------------- GEMM2: logits[1024,128] = X @ [cls_pos_w;cls_orient_w]^T ----------------
// Epilogue: bias + sigmoid, write probs to ws and scatter into pose_cls (interleaved).
__global__ void gemm2_wmma(const __bf16* __restrict__ X,   // [1024,1024] bf16
                           const __bf16* __restrict__ W,   // [128,1024] bf16
                           const float* __restrict__ pos_b,
                           const float* __restrict__ ori_b,
                           float* __restrict__ sig,        // ws [1024,128]
                           float* __restrict__ out_cls) {  // d_out + 1024*7
  const int lane = threadIdx.x & 31;
  const int wave = threadIdx.x >> 5;
  const int half = lane >> 4;
  const int lr   = lane & 15;
  const int row_base = blockIdx.x * 16;
  const int col_base = wave * 16;                 // 8 waves cover all 128 cols

  const __bf16* aRow = X + (long)(row_base + lr) * LATENT;
  const __bf16* bRow = W + (long)(col_base + lr) * LATENT;

  v8f c = {};
  #pragma unroll 4
  for (int kb = 0; kb < LATENT; kb += 32) {
    v8bf a0 = *reinterpret_cast<const v8bf*>(aRow + kb + half * 16);
    v8bf a1 = *reinterpret_cast<const v8bf*>(aRow + kb + half * 16 + 8);
    v8bf b0 = *reinterpret_cast<const v8bf*>(bRow + kb + half * 16);
    v8bf b1 = *reinterpret_cast<const v8bf*>(bRow + kb + half * 16 + 8);
    v16bf av = __builtin_shufflevector(a0, a1, 0,1,2,3,4,5,6,7,8,9,10,11,12,13,14,15);
    v16bf bv = __builtin_shufflevector(b0, b1, 0,1,2,3,4,5,6,7,8,9,10,11,12,13,14,15);
    c = __builtin_amdgcn_wmma_f32_16x16x32_bf16(false, av, false, bv, (short)0, c, false, false);
  }
  const int col = col_base + lr;                  // 0..127
  const float bcol = (col < 64) ? pos_b[col] : ori_b[col - 64];
  const int oi = (col < 64) ? (col * 2) : ((col - 64) * 2 + 1);
  #pragma unroll
  for (int i = 0; i < 8; ++i) {
    int row = row_base + i + half * 8;
    float z = c[i] + bcol;
    float s = 1.0f / (1.0f + __expf(-z));
    sig[(long)row * 128 + col] = s;
    out_cls[(long)row * 128 + oi] = s;
  }
}

// ---------------- ordinal decode + hard-routed expert matvecs ----------------
// One block (128 threads) per sample: leading-run label from sigmoid probs,
// then 7 dot products of length 1024 with LDS tree reduction (fp32).
__global__ void head_kernel(const float* __restrict__ sig, const float* __restrict__ Xf,
                            const float* __restrict__ rpw, const float* __restrict__ rpb,
                            const float* __restrict__ rqw, const float* __restrict__ rqb,
                            float* __restrict__ pose) {
  __shared__ int sidx[2];
  __shared__ float red[128];
  const int n = blockIdx.x, tid = threadIdx.x;
  if (tid == 0) {
    int r = 0;
    while (r < 64 && sig[(long)n * 128 + r] > 0.5f) ++r;
    sidx[0] = (r - 1 > 0) ? (r - 1) : 0;
    r = 0;
    while (r < 64 && sig[(long)n * 128 + 64 + r] > 0.5f) ++r;
    sidx[1] = (r - 1 > 0) ? (r - 1) : 0;
  }
  __syncthreads();
  const int ix = sidx[0], iq = sidx[1];
  const float* x = Xf + (long)n * LATENT;
  for (int o = 0; o < 7; ++o) {
    const float* w;
    float b;
    if (o < 3) { w = rpw + ((long)ix * 3 + o) * LATENT; b = rpb[ix * 3 + o]; }
    else       { int o2 = o - 3; w = rqw + ((long)iq * 4 + o2) * LATENT; b = rqb[iq * 4 + o2]; }
    float p = 0.f;
    for (int k = tid; k < LATENT; k += 128) p += w[k] * x[k];
    red[tid] = p;
    __syncthreads();
    for (int s = 64; s > 0; s >>= 1) {
      if (tid < s) red[tid] += red[tid + s];
      __syncthreads();
    }
    if (tid == 0) pose[(long)n * 7 + o] = red[0] + b;
    __syncthreads();
  }
}

extern "C" void kernel_launch(void* const* d_in, const int* in_sizes, int n_in,
                              void* d_out, int out_size, void* d_ws, size_t ws_size,
                              hipStream_t stream) {
  (void)in_sizes; (void)n_in; (void)out_size; (void)ws_size;
  const float* feat  = (const float*)d_in[0];
  const float* fc1_w = (const float*)d_in[1];
  const float* fc1_b = (const float*)d_in[2];
  const float* cpw   = (const float*)d_in[3];
  const float* cpb   = (const float*)d_in[4];
  const float* cow   = (const float*)d_in[5];
  const float* cob   = (const float*)d_in[6];
  const float* rpw   = (const float*)d_in[7];
  const float* rpb   = (const float*)d_in[8];
  const float* rqw   = (const float*)d_in[9];
  const float* rqb   = (const float*)d_in[10];
  float* out = (float*)d_out;

  char* ws = (char*)d_ws;
  __bf16* fc1w_b = (__bf16*)ws;  ws += (size_t)LATENT * BDIM * 2;     // 4 MB
  __bf16* pooled = (__bf16*)ws;  ws += (size_t)N_BATCH * BDIM * 2;    // 4 MB
  __bf16* clsw   = (__bf16*)ws;  ws += (size_t)128 * LATENT * 2;      // 256 KB
  float*  Xf     = (float*)ws;   ws += (size_t)N_BATCH * LATENT * 4;  // 4 MB
  __bf16* Xb     = (__bf16*)ws;  ws += (size_t)N_BATCH * LATENT * 2;  // 2 MB
  float*  sig    = (float*)ws;   ws += (size_t)N_BATCH * 128 * 4;     // 512 KB

  cvt_f32_bf16<<<(LATENT * BDIM + 255) / 256, 256, 0, stream>>>(fc1_w, fc1w_b, LATENT * BDIM);
  pack_clsw<<<(128 * LATENT + 255) / 256, 256, 0, stream>>>(cpw, cow, clsw);
  pool_kernel<<<(N_BATCH * BDIM) / 128, 256, 0, stream>>>(feat, pooled);
  gemm1_wmma<<<dim3(8, 64), 256, 0, stream>>>(pooled, fc1w_b, fc1_b, Xf, Xb);
  gemm2_wmma<<<64, 256, 0, stream>>>(Xb, clsw, cpb, cob, sig, out + (size_t)N_BATCH * 7);
  head_kernel<<<N_BATCH, 128, 0, stream>>>(sig, Xf, rpw, rpb, rqw, rqb, out);
}
